// JEPAPredictor_87883620811455
// MI455X (gfx1250) — matrix operand order repaired and checked by reading, not compiled, not run on hardware
//
#include <hip/hip_runtime.h>
#include <hip/hip_bf16.h>
#include <stdint.h>

// ---------------- problem constants ----------------
#define BB   4
#define TT   16
#define NTOK 128          // tokens per step (96 state + 32 action)
#define DM   1024
#define NH   16
#define DHD  64
#define LYR  4
#define MTOT (BB*TT*NTOK) // 8192 rows in the token stream

typedef __attribute__((ext_vector_type(8)))  __bf16 v8bf;
typedef __attribute__((ext_vector_type(16))) __bf16 v16bf;
typedef __attribute__((ext_vector_type(8)))  float  v8f;
typedef __attribute__((ext_vector_type(4)))  unsigned int u32x4;
typedef __attribute__((ext_vector_type(8)))  int i32x8;
typedef __attribute__((ext_vector_type(4)))  int i32x4;

#ifndef __has_builtin
#define __has_builtin(x) 0
#endif
#if defined(__gfx1250__) && __has_builtin(__builtin_amdgcn_tensor_load_to_lds) && \
    __has_builtin(__builtin_amdgcn_s_wait_tensorcnt)
#define USE_TDM 1
#else
#define USE_TDM 0
#endif

// ---------------- small helpers ----------------
__device__ __forceinline__ __bf16 f2bf(float f) {
    union { float f; unsigned u; } a{f};
    unsigned r = (a.u + 0x7FFFu + ((a.u >> 16) & 1u)) >> 16;
    union { unsigned short s; __bf16 b; } o{(unsigned short)r};
    return o.b;
}

__device__ __forceinline__ v16bf cat8(v8bf lo, v8bf hi) {
    return __builtin_shufflevector(lo, hi, 0,1,2,3,4,5,6,7,8,9,10,11,12,13,14,15);
}

__device__ __forceinline__ v8f wmma_bf16(v16bf a, v16bf b, v8f c) {
    // D = A(16x32 bf16) * B(32x16 bf16) + C(16x16 f32)
    return __builtin_amdgcn_wmma_f32_16x16x32_bf16(false, a, false, b, (short)0, c, false, false);
}

__device__ __forceinline__ float gelu_tanh(float x) {
    float x3 = x * x * x;
    return 0.5f * x * (1.0f + tanhf(0.7978845608f * (x + 0.044715f * x3)));
}

// ---------------- elementwise kernels ----------------
__global__ __launch_bounds__(256) void concat_kernel(const float* __restrict__ s,
                                                     const float* __restrict__ a,
                                                     float* __restrict__ x) {
    size_t i = (size_t)blockIdx.x * 256 + threadIdx.x;   // over MTOT*DM
    int c   = (int)(i & (DM - 1));
    int row = (int)(i >> 10);
    int tok = row & (NTOK - 1);
    int bt  = row >> 7;
    float v = (tok < 96) ? s[((size_t)bt * 96 + tok) * DM + c]
                         : a[((size_t)bt * 32 + (tok - 96)) * DM + c];
    x[i] = v;
}

__global__ __launch_bounds__(256) void cvt_kernel(const float* __restrict__ src,
                                                  __bf16* __restrict__ dst, unsigned n) {
    unsigned i = blockIdx.x * 256 + threadIdx.x;
    if (i < n) dst[i] = f2bf(src[i]);
}

__global__ __launch_bounds__(256) void rmsnorm_kernel(const float* __restrict__ x,
                                                      const float* __restrict__ w,
                                                      __bf16* __restrict__ out) {
    int row = blockIdx.x;
    const float* xr = x + (size_t)row * DM;
    float s = 0.f;
    for (int i = threadIdx.x; i < DM; i += 256) { float v = xr[i]; s += v * v; }
    s += __shfl_xor(s, 1);  s += __shfl_xor(s, 2);
    s += __shfl_xor(s, 4);  s += __shfl_xor(s, 8);  s += __shfl_xor(s, 16);
    __shared__ float red[8];
    if ((threadIdx.x & 31) == 0) red[threadIdx.x >> 5] = s;
    __syncthreads();
    float tot = red[0] + red[1] + red[2] + red[3] + red[4] + red[5] + red[6] + red[7];
    float rs = rsqrtf(tot * (1.0f / DM) + 1e-5f);
    for (int i = threadIdx.x; i < DM; i += 256)
        out[(size_t)row * DM + i] = f2bf(xr[i] * rs * w[i]);
}

// ---------------- bf16 WMMA GEMM: out[M,N] = act(A[M,K] @ W[N,K]^T) (+resid) ----------------
// Block tile 128x128, 8 waves (4 along M x 2 along N), wave tile 32x64, K step 32.
// A tile staged to LDS (80B row pitch) via TDM (double-buffered) when available.
#define TILEB (128 * 80)

template <int OUT_BF, int ACT_GELU, int RESID, int VSTORE>
__global__ __launch_bounds__(256) void gemm_kernel(const __bf16* __restrict__ A,
                                                   const __bf16* __restrict__ W,
                                                   const float* resid, void* outp,
                                                   int M, int N, int K) {
    __shared__ __align__(16) char smem[2 * TILEB];
    const int tid = threadIdx.x, lane = tid & 31, w = tid >> 5;
    const int wm = w & 3, wn = w >> 2;
    const int bm = blockIdx.y * 128, bn = blockIdx.x * 128;
    const int nkb = K >> 5;
    const int lm = lane & 15, hf = (lane >> 4) & 1;

    __builtin_prefetch(W + (size_t)bn * K, 0, 0);  // warm L2 for the weight panel

    v8f acc[2][4] = {};

    auto issue_tile = [&](int kb, int buf) {
#if USE_TDM
        if (tid < 32) {  // wave 0 drives the Tensor Data Mover
            unsigned lds_addr = (unsigned)(size_t)(&smem[buf * TILEB]);
            unsigned long long ga =
                (unsigned long long)(size_t)(A + (size_t)bm * K + (size_t)kb * 32);
            u32x4 g0;
            g0[0] = 1u;                      // count=1, gather off
            g0[1] = lds_addr;                // D#.lds_addr
            g0[2] = (unsigned)ga;            // D#.global_addr[31:0]
            g0[3] = (unsigned)((ga >> 32) & 0x1FFFFFFu) | (2u << 30);  // addr[56:32] | type=2
            i32x8 g1;
            // data_size=2B, pad_enable, pad_interval=16 DWORDs(64B row), pad_amount=4 DWORDs(16B)
            g1[0] = (int)((1u << 16) | (1u << 20) | (3u << 22) | (3u << 25));
            g1[1] = (int)(((unsigned)K & 0xFFFFu) << 16);                    // tensor_dim0 lo
            g1[2] = (int)(((unsigned)K >> 16) | (((unsigned)M & 0xFFFFu) << 16));
            g1[3] = (int)(((unsigned)M >> 16) | (32u << 16));                // tile_dim0 = 32
            g1[4] = 128;                                                     // tile_dim1 = 128
            g1[5] = (int)K;                                                  // dim0_stride lo
            g1[6] = 0; g1[7] = 0;
            i32x4 gz4 = {0, 0, 0, 0};
            i32x8 gz8 = {0, 0, 0, 0, 0, 0, 0, 0};
            __builtin_amdgcn_tensor_load_to_lds(g0, g1, gz4, gz4, gz8, 0);
        }
#else
        // cooperative fallback: 512 x 16B chunks, 2 per thread
        int cid = tid * 2;
        #pragma unroll
        for (int r = 0; r < 2; ++r) {
            int c = cid + r;
            int row = c >> 2, ch = c & 3;
            const v8bf* g = (const v8bf*)(A + (size_t)(bm + row) * K + (size_t)kb * 32 + ch * 8);
            *(v8bf*)(&smem[buf * TILEB + row * 80 + ch * 16]) = *g;
        }
#endif
    };
    auto wait_tile = [&]() {
#if USE_TDM
        if (tid < 32) __builtin_amdgcn_s_wait_tensorcnt(0);
#endif
        __syncthreads();
    };

    issue_tile(0, 0);
    wait_tile();

    // per-thread loop-invariant bases; all inner addresses = base + small constant
    const char* abase = smem + (wm * 32 + lm) * 80 + hf * 16;       // A frag base (buffer 0)
    const char* wbase = (const char*)W + 2 * ((size_t)(bn + wn * 64 + lm) * K) + hf * 16;
    const size_t ntoff = 32 * (size_t)K;                             // bytes per 16-row weight tile

    for (int kb = 0; kb < nkb; ++kb) {
        const int cur = kb & 1;
        if (kb + 1 < nkb) issue_tile(kb + 1, cur ^ 1);

        const char* ta = abase + cur * TILEB;
        v16bf af0 = cat8(*(const v8bf*)ta, *(const v8bf*)(ta + 32));
        v16bf af1 = cat8(*(const v8bf*)(ta + 16 * 80), *(const v8bf*)(ta + 16 * 80 + 32));
        const char* wp = wbase + (size_t)kb * 64;
        #pragma unroll
        for (int nt = 0; nt < 4; ++nt) {
            v16bf bf_ = cat8(*(const v8bf*)(wp + nt * ntoff), *(const v8bf*)(wp + nt * ntoff + 32));
            acc[0][nt] = wmma_bf16(af0, bf_, acc[0][nt]);
            acc[1][nt] = wmma_bf16(af1, bf_, acc[1][nt]);
        }
        wait_tile();
    }

    // epilogue: C element (vgpr i, lane) -> row = i + 8*(lane>>4), col = lane&15
    const int rofs = 8 * hf;
    #pragma unroll
    for (int mt = 0; mt < 2; ++mt) {
        #pragma unroll
        for (int i = 0; i < 8; ++i) {
            const int m = bm + wm * 32 + mt * 16 + i + rofs;
            const size_t rowb = (size_t)m * N + bn + wn * 64 + lm;  // stepped by strength reduction
            #pragma unroll
            for (int nt = 0; nt < 4; ++nt) {
                const int n = bn + wn * 64 + nt * 16 + lm;
                float v = acc[mt][nt][i];
                if (ACT_GELU) v = gelu_tanh(v);
                if (RESID) v += resid[rowb + nt * 16];
                if (OUT_BF) {
                    __bf16* o = (__bf16*)outp;
                    if (VSTORE) {  // V stored transposed: [bt][h][dh][tok]
                        int bt = m >> 7, tok = m & 127, hh = n >> 6, dh = n & 63;
                        o[(((size_t)bt * NH + hh) * DHD + dh) * NTOK + tok] = f2bf(v);
                    } else {
                        o[rowb + nt * 16] = f2bf(v);
                    }
                } else {
                    ((float*)outp)[rowb + nt * 16] = v;
                }
            }
        }
    }
}

// ---------------- fused banded attention ----------------
// grid = (B*T, H), 256 threads = 8 waves; wave w owns queries [w*16, w*16+16)
// keys = 2N band: tiles 0..7 = step t-1, tiles 8..15 = step t
__global__ __launch_bounds__(256) void attn_kernel(const __bf16* __restrict__ Q,
                                                   const __bf16* __restrict__ Kb,
                                                   const __bf16* __restrict__ VT,
                                                   const unsigned char* __restrict__ tmask,
                                                   __bf16* __restrict__ outp) {
    __shared__ __align__(16) __bf16 probs[8][16][256];  // 64KB
    const int tid = threadIdx.x, lane = tid & 31, w = tid >> 5;
    const int bt = blockIdx.x, h = blockIdx.y;
    const int t = bt & (TT - 1);
    const int half = lane >> 4, lm = lane & 15;
    const bool stepOK = tmask[bt] != 0;
    const bool prevOK = (t > 0) && (tmask[bt - 1] != 0);
    const int btPrev = (t > 0) ? bt - 1 : bt;

    // Q A-fragments (Dh = 64 -> two K=32 windows); base + {0,16} / {64,80} bytes
    const char* qbase = (const char*)Q +
        2 * (((size_t)(bt * NTOK + w * 16 + lm)) * DM + h * DHD + half * 16);
    v16bf aq[2];
    aq[0] = cat8(*(const v8bf*)qbase, *(const v8bf*)(qbase + 16));
    aq[1] = cat8(*(const v8bf*)(qbase + 64), *(const v8bf*)(qbase + 80));

    // scores = Q @ K^T over 16 key tiles; key-tile stride = 16 rows * 2048B = 32768B (imm offset)
    const char* kbase[2];
    kbase[0] = (const char*)Kb + 2 * (((size_t)btPrev * NTOK + lm) * DM + h * DHD + half * 16);
    kbase[1] = (const char*)Kb + 2 * (((size_t)bt     * NTOK + lm) * DM + h * DHD + half * 16);
    v8f S[16];
    #pragma unroll
    for (int nt = 0; nt < 16; ++nt) {
        const char* kp = kbase[nt >> 3] + (nt & 7) * 32768;
        v8f a = {};
        a = wmma_bf16(aq[0], *(const v16bf*)kp, a);
        a = wmma_bf16(aq[1], *(const v16bf*)(kp + 64), a);
        S[nt] = a;
    }

    // masked softmax over 256 keys; row = i + 8*half lives in one 16-lane half
    const float scale = 0.125f;  // 1/sqrt(64)
    float rinv[8], rmax[8];
    #pragma unroll
    for (int i = 0; i < 8; ++i) {
        float mx = -3.0e38f;
        #pragma unroll
        for (int nt = 0; nt < 16; ++nt) {
            bool ok = (nt < 8) ? prevOK : stepOK;
            float v = ok ? S[nt][i] * scale : -1.0e30f;
            S[nt][i] = v;
            mx = fmaxf(mx, v);
        }
        mx = fmaxf(mx, __shfl_xor(mx, 1));
        mx = fmaxf(mx, __shfl_xor(mx, 2));
        mx = fmaxf(mx, __shfl_xor(mx, 4));
        mx = fmaxf(mx, __shfl_xor(mx, 8));
        rmax[i] = mx;
    }
    #pragma unroll
    for (int i = 0; i < 8; ++i) {
        float s = 0.f;
        #pragma unroll
        for (int nt = 0; nt < 16; ++nt) {
            float e = __expf(S[nt][i] - rmax[i]);
            S[nt][i] = e;
            s += e;
        }
        s += __shfl_xor(s, 1); s += __shfl_xor(s, 2);
        s += __shfl_xor(s, 4); s += __shfl_xor(s, 8);
        rinv[i] = 1.0f / s;
    }
    // probs -> LDS (per-wave strip, row pitch 512B); all offsets are immediates
    {
        char* psto = (char*)&probs[w][0][0] + (8 * half) * 512 + lm * 2;
        #pragma unroll
        for (int nt = 0; nt < 16; ++nt)
            #pragma unroll
            for (int i = 0; i < 8; ++i)
                *(__bf16*)(psto + i * 512 + nt * 32) = f2bf(S[nt][i] * rinv[i]);
    }
    // only this wave reads its strip -> no block barrier needed (DS counters serialize)

    // attn = probs @ Vband ; V pre-transposed: VT[bt][h][dh][tok]
    const char* pbase = (const char*)&probs[w][lm][0] + half * 32;
    const char* vbase[2];
    vbase[0] = (const char*)VT + 2 * ((((size_t)btPrev * NH + h) * DHD + lm) * NTOK) + half * 32;
    vbase[1] = (const char*)VT + 2 * ((((size_t)bt     * NH + h) * DHD + lm) * NTOK) + half * 32;
    v8f acc[4] = {};
    #pragma unroll
    for (int kk = 0; kk < 8; ++kk) {
        v16bf ap = cat8(*(const v8bf*)(pbase + kk * 64), *(const v8bf*)(pbase + kk * 64 + 16));
        const char* vb = vbase[kk >> 2] + (kk & 3) * 64;  // 32 tokens * 2B per k-step
        #pragma unroll
        for (int nt = 0; nt < 4; ++nt)  // dh tile stride = 16*NTOK*2 = 4096B
            acc[nt] = wmma_bf16(ap, *(const v16bf*)(vb + nt * 4096), acc[nt]);
    }

    const float zm = stepOK ? 1.0f : 0.0f;
    char* obase = (char*)outp +
        2 * (((size_t)(bt * NTOK + w * 16 + 8 * half)) * DM + h * DHD + lm);
    #pragma unroll
    for (int nt = 0; nt < 4; ++nt)
        #pragma unroll
        for (int i = 0; i < 8; ++i)
            *(__bf16*)(obase + i * (2 * DM) + nt * 32) = f2bf(acc[nt][i] * zm);
}

// ---------------- host driver ----------------
extern "C" void kernel_launch(void* const* d_in, const int* in_sizes, int n_in,
                              void* d_out, int out_size, void* d_ws, size_t ws_size,
                              hipStream_t stream) {
    const float* state  = (const float*)d_in[0];
    const float* action = (const float*)d_in[1];
    const unsigned char* tmask = (const unsigned char*)d_in[2];
    const float* norm_w = (const float*)d_in[3];
    const float* wq = (const float*)d_in[4];
    const float* wk = (const float*)d_in[5];
    const float* wv = (const float*)d_in[6];
    const float* wo = (const float*)d_in[7];
    const float* ffn_norm_w = (const float*)d_in[8];
    const float* w1 = (const float*)d_in[9];
    const float* w2 = (const float*)d_in[10];
    float* x = (float*)d_out;  // residual stream lives in d_out (in-place updates are 1:1)

    char* ws = (char*)d_ws;
    size_t off = 0;
    auto alloc = [&](size_t nelem) -> __bf16* {
        __bf16* p = (__bf16*)(ws + off);
        off = (off + nelem * 2 + 255) & ~(size_t)255;
        return p;
    };
    const size_t dd = (size_t)DM * DM;
    __bf16* wq_b = alloc((size_t)LYR * dd);
    __bf16* wk_b = alloc((size_t)LYR * dd);
    __bf16* wv_b = alloc((size_t)LYR * dd);
    __bf16* wo_b = alloc((size_t)LYR * dd);
    __bf16* w1_b = alloc((size_t)LYR * 4 * dd);
    __bf16* w2_b = alloc((size_t)LYR * 4 * dd);
    __bf16* hbuf = alloc((size_t)MTOT * DM);
    __bf16* qb   = alloc((size_t)MTOT * DM);
    __bf16* kbuf = alloc((size_t)MTOT * DM);
    __bf16* vtb  = alloc((size_t)MTOT * DM);
    __bf16* ab   = alloc((size_t)MTOT * DM);
    __bf16* f1   = alloc((size_t)MTOT * 4 * DM);

    // x = concat(state, action)
    concat_kernel<<<(MTOT * (size_t)DM) / 256, 256, 0, stream>>>(state, action, x);

    // one-shot weight quantization to bf16
    auto cvt = [&](const float* s, __bf16* d, size_t n) {
        cvt_kernel<<<(unsigned)((n + 255) / 256), 256, 0, stream>>>(s, d, (unsigned)n);
    };
    cvt(wq, wq_b, LYR * dd); cvt(wk, wk_b, LYR * dd);
    cvt(wv, wv_b, LYR * dd); cvt(wo, wo_b, LYR * dd);
    cvt(w1, w1_b, LYR * 4 * dd); cvt(w2, w2_b, LYR * 4 * dd);

    dim3 gSq(DM / 128, MTOT / 128);       // [8192 x 1024] projections
    dim3 gF1(4 * DM / 128, MTOT / 128);   // [8192 x 4096] FFN up

    for (int l = 0; l < LYR; ++l) {
        // attention block
        rmsnorm_kernel<<<MTOT, 256, 0, stream>>>(x, norm_w + (size_t)l * DM, hbuf);
        gemm_kernel<1, 0, 0, 0><<<gSq, 256, 0, stream>>>(hbuf, wq_b + l * dd, nullptr, qb,  MTOT, DM, DM);
        gemm_kernel<1, 0, 0, 0><<<gSq, 256, 0, stream>>>(hbuf, wk_b + l * dd, nullptr, kbuf, MTOT, DM, DM);
        gemm_kernel<1, 0, 0, 1><<<gSq, 256, 0, stream>>>(hbuf, wv_b + l * dd, nullptr, vtb, MTOT, DM, DM);
        attn_kernel<<<dim3(BB * TT, NH), 256, 0, stream>>>(qb, kbuf, vtb, tmask, ab);
        gemm_kernel<0, 0, 1, 0><<<gSq, 256, 0, stream>>>(ab, wo_b + l * dd, x, x, MTOT, DM, DM);
        // FFN block
        rmsnorm_kernel<<<MTOT, 256, 0, stream>>>(x, ffn_norm_w + (size_t)l * DM, hbuf);
        gemm_kernel<1, 1, 0, 0><<<gF1, 256, 0, stream>>>(hbuf, w1_b + (size_t)l * 4 * dd, nullptr, f1, MTOT, 4 * DM, DM);
        gemm_kernel<0, 0, 1, 0><<<gSq, 256, 0, stream>>>(f1, w2_b + (size_t)l * 4 * dd, x, x, MTOT, DM, 4 * DM);
    }
}
